// FPModule_17154099380546
// MI455X (gfx1250) — matrix operand order repaired and checked by reading, not compiled, not run on hardware
//
#include <hip/hip_runtime.h>
#include <hip/hip_bf16.h>

// Problem constants (match reference).
#define B_    8
#define NS_   2048
#define NT_   8192
#define CIN_  256
#define CSK_  128
#define H_    256
#define COUT_ 128
#define M_    (B_ * NT_)        // 65536 rows through the MLP
#define KD1_  (CIN_ + CSK_)     // 384 = first GEMM K

typedef __bf16 bf16_t;
typedef bf16_t v16bf  __attribute__((ext_vector_type(16)));
typedef bf16_t bf16x8 __attribute__((ext_vector_type(8)));
typedef bf16_t bf16x4 __attribute__((ext_vector_type(4)));
typedef float  v8f    __attribute__((ext_vector_type(8)));
typedef unsigned int u32x4 __attribute__((ext_vector_type(4)));
typedef int          i32x4 __attribute__((ext_vector_type(4)));
typedef int          i32x8 __attribute__((ext_vector_type(8)));

// ---------------------------------------------------------------------------
// Weight prep: fp32 [K][N] -> bf16 [N][K] (transposed so WMMA B-fragments are
// contiguous-K loads per lane).
// ---------------------------------------------------------------------------
__global__ __launch_bounds__(256) void convert_transpose_kernel(
    const float* __restrict__ W, bf16_t* __restrict__ WT, int K, int N) {
  int idx = blockIdx.x * 256 + threadIdx.x;
  if (idx >= K * N) return;
  int k = idx / N, n = idx - k * N;
  WT[(size_t)n * K + k] = (bf16_t)W[(size_t)k * N + n];
}

// ---------------------------------------------------------------------------
// Fused kNN(3) + inverse-distance interpolate + concat(x_skip) -> bf16 hbuf.
// One thread per target point; 2048 sources (pos + |s|^2 as float4) staged in
// 32KB LDS; uniform LDS address per iteration => broadcast read.
// ---------------------------------------------------------------------------
__global__ __launch_bounds__(256) void knn_interp_kernel(
    const float* __restrict__ x, const float* __restrict__ pos,
    const float* __restrict__ x_skip, const float* __restrict__ pos_skip,
    bf16_t* __restrict__ hbuf) {
  __shared__ float4 sp[NS_];
  const int b   = blockIdx.y;
  const int tid = threadIdx.x;

  for (int j = tid; j < NS_; j += 256) {
    const float* p = pos + (size_t)(b * NS_ + j) * 3;
    float sx = p[0], sy = p[1], sz = p[2];
    sp[j] = make_float4(sx, sy, sz, sx * sx + sy * sy + sz * sz);
  }
  __syncthreads();

  const int t = b * NT_ + blockIdx.x * 256 + tid;
  const float* q = pos_skip + (size_t)t * 3;
  const float tx = q[0], ty = q[1], tz = q[2];
  const float tn2 = tx * tx + ty * ty + tz * tz;

  float d0 = 3.4e38f, d1 = 3.4e38f, d2v = 3.4e38f;
  int   i0 = 0, i1 = 0, i2 = 0;
#pragma unroll 4
  for (int j = 0; j < NS_; ++j) {
    float4 s  = sp[j];
    float  dt = fmaf(tx, s.x, fmaf(ty, s.y, tz * s.z));
    float  d  = fmaf(-2.0f, dt, tn2 + s.w);
    if (d < d2v) {                 // rare: wave branches over update
      if (d < d1) {
        if (d < d0) { d2v = d1; i2 = i1; d1 = d0; i1 = i0; d0 = d; i0 = j; }
        else        { d2v = d1; i2 = i1; d1 = d;  i1 = j; }
      } else        { d2v = d;  i2 = j; }
    }
  }

  float w0 = 1.0f / fmaxf(d0, 1e-16f);
  float w1 = 1.0f / fmaxf(d1, 1e-16f);
  float w2 = 1.0f / fmaxf(d2v, 1e-16f);
  float inv = 1.0f / (w0 + w1 + w2);
  w0 *= inv; w1 *= inv; w2 *= inv;

  const float4* r0 = (const float4*)(x + (size_t)(b * NS_ + i0) * CIN_);
  const float4* r1 = (const float4*)(x + (size_t)(b * NS_ + i1) * CIN_);
  const float4* r2 = (const float4*)(x + (size_t)(b * NS_ + i2) * CIN_);
  bf16_t* hr = hbuf + (size_t)t * KD1_;
#pragma unroll 4
  for (int c = 0; c < CIN_ / 4; ++c) {
    float4 a = r0[c], bb = r1[c], cc = r2[c];
    bf16x4 o;
    o.x = (bf16_t)(w0 * a.x + w1 * bb.x + w2 * cc.x);
    o.y = (bf16_t)(w0 * a.y + w1 * bb.y + w2 * cc.y);
    o.z = (bf16_t)(w0 * a.z + w1 * bb.z + w2 * cc.z);
    o.w = (bf16_t)(w0 * a.w + w1 * bb.w + w2 * cc.w);
    *(bf16x4*)(hr + c * 4) = o;
  }
  const float4* rs = (const float4*)(x_skip + (size_t)t * CSK_);
  bf16_t* hs = hr + CIN_;
#pragma unroll 4
  for (int c = 0; c < CSK_ / 4; ++c) {
    float4 a = rs[c];
    bf16x4 o;
    o.x = (bf16_t)a.x; o.y = (bf16_t)a.y; o.z = (bf16_t)a.z; o.w = (bf16_t)a.w;
    *(bf16x4*)(hs + c * 4) = o;
  }
}

// ---------------------------------------------------------------------------
// WMMA GEMM with TDM-staged weights:
//   out[M][ND] = act(A[M][KD](bf16) @ BT[ND][KD]^T(bf16) + bias)
// Block = 8 waves = 128 rows. K is the outer loop in KCHUNK=128 chunks; the
// weight tile BT[0:ND][kb:kb+128] (<=64KB) is DMA'd once per block per chunk
// into LDS by the Tensor Data Mover, then all 8 waves feed WMMA B-fragments
// from LDS while each wave keeps all ND/16 accumulators live in VGPRs and
// reuses each A fragment across every N tile (A is read exactly once).
// Fragment layouts used here (contiguous k, same order for A and B):
//   A lane L: row m=L&15; elems = K[hi*16 .. +15] (contiguous in A row)
//   B lane L: col n=L&15; elems = K[hi*16 .. +15] (contiguous in BT row)
//   C/D lane L: col n=L&15; VGPR r holds row m = r + 8*hi
// ---------------------------------------------------------------------------
template <int KD, int ND, bool RELU, bool OUTBF>
__global__ __launch_bounds__(256) void wmma_mlp_gemm_kernel(
    const bf16_t* __restrict__ A, const bf16_t* __restrict__ BT,
    const float* __restrict__ bias, void* __restrict__ outp) {
  constexpr int KCHUNK = 128;
  constexpr int KSTEPS = KCHUNK / 32;   // WMMAs per chunk per N tile
  constexpr int NTILES = ND / 16;
  __shared__ bf16_t btile[ND * KCHUNK]; // layer1: 64KB, layer2: 32KB

  const int lane = threadIdx.x & 31;
  const int wave = threadIdx.x >> 5;
  const int m0   = (blockIdx.x * 8 + wave) * 16;
  const int mr   = lane & 15;
  const int hi   = lane >> 4;

  v8f acc[NTILES];
#pragma unroll
  for (int nt = 0; nt < NTILES; ++nt) acc[nt] = v8f{};

  const unsigned lds_base = (unsigned)(uintptr_t)(&btile[0]);

  for (int kb = 0; kb < KD; kb += KCHUNK) {
    __syncthreads();  // previous chunk fully consumed before overwrite

#if __has_builtin(__builtin_amdgcn_tensor_load_to_lds)
    if (threadIdx.x < 32) {
      // Tensor DMA descriptor (CDNA5 ISA):
      // 2D tile: dim0 = KCHUNK bf16 elems (contiguous), dim1 = ND rows,
      // global row stride = KD elems, destination = btile.
      const unsigned long long ga =
          (unsigned long long)(uintptr_t)(BT + kb);
      u32x4 g0;
      g0[0] = 0x1u;                                  // count=1, user mode
      g0[1] = lds_base;                              // lds_addr
      g0[2] = (unsigned)(ga & 0xffffffffu);          // global_addr[31:0]
      g0[3] = (unsigned)((ga >> 32) & 0x01ffffffu)   // global_addr[56:32]
              | (2u << 30);                          // D# type = 2 (image)
      i32x8 g1;
      g1[0] = (int)(1u << 16);                       // data_size=1 -> 2 bytes
      g1[1] = (int)(((unsigned)KCHUNK & 0xffffu) << 16);         // tensor_dim0 lo16 (bits 63:48)
      g1[2] = (int)((((unsigned)KCHUNK >> 16) & 0xffffu)         // tensor_dim0 hi16
              | (((unsigned)ND & 0xffffu) << 16));               // tensor_dim1 lo16
      g1[3] = (int)((((unsigned)ND >> 16) & 0xffffu)             // tensor_dim1 hi16
              | (((unsigned)KCHUNK & 0xffffu) << 16));           // tile_dim0
      g1[4] = (int)((unsigned)ND & 0xffffu);                     // tile_dim1 (tile_dim2=0)
      g1[5] = (int)(unsigned)KD;                                 // tensor_dim0_stride lo32
      g1[6] = 0;                                                 // stride hi16 | dim1_stride lo
      g1[7] = 0;
      i32x4 z4 = {};
#if __clang_major__ >= 23
      i32x8 z8 = {};
      __builtin_amdgcn_tensor_load_to_lds(g0, g1, z4, z4, z8, 0);
#else
      __builtin_amdgcn_tensor_load_to_lds(g0, g1, z4, z4, 0);
#endif
      __builtin_amdgcn_s_wait_tensorcnt(0);  // only issuing wave's TENSORcnt
    }
#else
    // Fallback: cooperative copy, 16B per thread per iteration.
    for (int i = threadIdx.x; i < ND * (KCHUNK / 8); i += 256) {
      int n = i / (KCHUNK / 8), c8 = i - n * (KCHUNK / 8);
      *(bf16x8*)(btile + n * KCHUNK + c8 * 8) =
          *(const bf16x8*)(BT + (size_t)n * KD + kb + c8 * 8);
    }
#endif
    __syncthreads();  // publish LDS tile to all 8 waves

    const bf16_t* arow = A + (size_t)(m0 + mr) * KD + kb;
#pragma unroll
    for (int ks = 0; ks < KSTEPS; ++ks) {
      bf16x8 lo = *(const bf16x8*)(arow + ks * 32 + hi * 16);
      bf16x8 h8 = *(const bf16x8*)(arow + ks * 32 + hi * 16 + 8);
      v16bf afrag = __builtin_shufflevector(lo, h8, 0, 1, 2, 3, 4, 5, 6, 7,
                                            8, 9, 10, 11, 12, 13, 14, 15);
#pragma unroll
      for (int nt = 0; nt < NTILES; ++nt) {
        v16bf bfr = *(const v16bf*)(btile + (nt * 16 + mr) * KCHUNK +
                                    ks * 32 + hi * 16);
        acc[nt] = __builtin_amdgcn_wmma_f32_16x16x32_bf16(
            /*neg_a=*/false, afrag, /*neg_b=*/false, bfr,
            /*c_mod=*/(short)0, acc[nt], /*reuse_a=*/false, /*reuse_b=*/false);
      }
    }
  }

  // Epilogue: bias (+ReLU), store 16 x ND strip.
#pragma unroll
  for (int nt = 0; nt < NTILES; ++nt) {
    const float bv = bias[nt * 16 + mr];
#pragma unroll
    for (int r = 0; r < 8; ++r) {
      float v = acc[nt][r] + bv;
      if (RELU) v = fmaxf(v, 0.0f);
      const size_t o = (size_t)(m0 + r + 8 * hi) * ND + nt * 16 + mr;
      if (OUTBF) ((bf16_t*)outp)[o] = (bf16_t)v;
      else       ((float*)outp)[o]  = v;
    }
  }
}

// ---------------------------------------------------------------------------
extern "C" void kernel_launch(void* const* d_in, const int* in_sizes, int n_in,
                              void* d_out, int out_size, void* d_ws, size_t ws_size,
                              hipStream_t stream) {
  const float* x        = (const float*)d_in[0];
  const float* pos      = (const float*)d_in[1];
  const float* x_skip   = (const float*)d_in[3];
  const float* pos_skip = (const float*)d_in[4];
  const float* W1       = (const float*)d_in[6];
  const float* b1       = (const float*)d_in[7];
  const float* W2       = (const float*)d_in[8];
  const float* b2       = (const float*)d_in[9];

  // Workspace layout (bf16): hbuf[M][384] | h1[M][256] | W1T[256][384] | W2T[128][256]
  char*  ws   = (char*)d_ws;
  size_t off  = 0;
  bf16_t* hbuf = (bf16_t*)(ws + off); off += (size_t)M_ * KD1_ * sizeof(bf16_t);
  bf16_t* h1   = (bf16_t*)(ws + off); off += (size_t)M_ * H_ * sizeof(bf16_t);
  bf16_t* w1t  = (bf16_t*)(ws + off); off += (size_t)KD1_ * H_ * sizeof(bf16_t);
  bf16_t* w2t  = (bf16_t*)(ws + off);

  convert_transpose_kernel<<<(KD1_ * H_ + 255) / 256, 256, 0, stream>>>(W1, w1t, KD1_, H_);
  convert_transpose_kernel<<<(H_ * COUT_ + 255) / 256, 256, 0, stream>>>(W2, w2t, H_, COUT_);

  knn_interp_kernel<<<dim3(NT_ / 256, B_), 256, 0, stream>>>(x, pos, x_skip, pos_skip, hbuf);

  wmma_mlp_gemm_kernel<KD1_, H_, true, true>
      <<<M_ / 128, 256, 0, stream>>>(hbuf, w1t, b1, h1);
  wmma_mlp_gemm_kernel<H_, COUT_, false, false>
      <<<M_ / 128, 256, 0, stream>>>(h1, w2t, b2, d_out);

  // Tuple outputs 2 & 3 are pass-throughs: pos_skip, batch_skip.
  const size_t main_elems = (size_t)M_ * COUT_;
  if ((size_t)out_size >= main_elems + (size_t)in_sizes[4] + (size_t)in_sizes[5]) {
    float* outf = (float*)d_out;
    hipMemcpyAsync(outf + main_elems, d_in[4],
                   (size_t)in_sizes[4] * sizeof(float),
                   hipMemcpyDeviceToDevice, stream);
    hipMemcpyAsync(outf + main_elems + in_sizes[4], d_in[5],
                   (size_t)in_sizes[5] * sizeof(int),
                   hipMemcpyDeviceToDevice, stream);
  }
}
